// Net_44281112822487
// MI455X (gfx1250) — compile-verified, matrix operand order checked by reading, not run on hardware
//
#include <hip/hip_runtime.h>
#include <hip/hip_bf16.h>
#include <math.h>

// ---------------------------------------------------------------------------
// Graph U-Net forward for MI455X (gfx1250, wave32).
//  - Edge message passing: gather + global_atomic_add_f32 scatter (L2-resident)
//  - Dense linears: V_WMMA_F32_16X16X4_F32 (fp32 matrix core; f16 WMMA fallback)
//  - TopK pooling: radix-select threshold + atomic compaction (order-free)
// ---------------------------------------------------------------------------

typedef __attribute__((ext_vector_type(2)))  float    v2f;
typedef __attribute__((ext_vector_type(8)))  float    v8f;
typedef __attribute__((ext_vector_type(16))) _Float16 v16h;

#define HAVE_WMMA_F32 __has_builtin(__builtin_amdgcn_wmma_f32_16x16x4_f32)

// ---------------------------------------------------------------------------
// One 16x16 WMMA tile accumulation: c += A_tile(16xCIN) * W_tile(16xCIN)^T
// arow points at A row (m0 + lane%16), wrow at W row (n0 + lane%16).
// ---------------------------------------------------------------------------
template<int CIN>
__device__ inline v8f gemm_tile(const float* __restrict__ arow,
                                const float* __restrict__ wrow,
                                int half, v8f c)
{
#if HAVE_WMMA_F32
  // fp32 WMMA 16x16x4.  A layout: lanes 0-15 hold K=k,k+1; lanes 16-31 K=k+2,k+3.
  #pragma unroll
  for (int k = 0; k < CIN; k += 4) {
    v2f a, b;
    a.x = arow[k + 2 * half];
    a.y = arow[k + 2 * half + 1];
    b.x = wrow[k + 2 * half];
    b.y = wrow[k + 2 * half + 1];
    c = __builtin_amdgcn_wmma_f32_16x16x4_f32(false, a, false, b,
                                              (short)0, c, false, false);
  }
#else
  // Fallback: codegen-confirmed f16 WMMA 16x16x32 (fp32 accumulate).
  // 16-bit A layout: lane<16: elems0-7=K0-7, elems8-15=K16-23; lane>=16: +8.
  #pragma unroll
  for (int k = 0; k < CIN; k += 32) {
    v16h a, b;
    #pragma unroll
    for (int j = 0; j < 8; ++j) {
      a[j]     = (_Float16)arow[k + 8 * half + j];
      a[8 + j] = (_Float16)arow[k + 16 + 8 * half + j];
      b[j]     = (_Float16)wrow[k + 8 * half + j];
      b[8 + j] = (_Float16)wrow[k + 16 + 8 * half + j];
    }
    c = __builtin_amdgcn_wmma_f32_16x16x32_f16(false, a, false, b,
                                               (short)0, c, false, false);
  }
#endif
  return c;
}

// out[M,Cout] = relu( agg[M,CIN] @ wrel^T + xr[M,CIN] @ wroot^T + bias )
// One wave (32 threads) per 16x16 output tile.
template<int CIN>
__global__ void graphconv_gemm(const float* __restrict__ agg,
                               const float* __restrict__ xr,
                               const float* __restrict__ wrel,
                               const float* __restrict__ wroot,
                               const float* __restrict__ bias,
                               float* __restrict__ out,
                               int M, int Cout)
{
  const int m0   = blockIdx.x * 16;
  const int n0   = blockIdx.y * 16;
  const int lane = threadIdx.x & 31;
  const int half = lane >> 4;
  const int r    = lane & 15;

  int mr = m0 + r;
  if (mr >= M) mr = M - 1;   // safe clamp (M is a multiple of 16 here)

  v8f c = {};
  c = gemm_tile<CIN>(agg + (size_t)mr * CIN,        wrel  + (size_t)(n0 + r) * CIN, half, c);
  c = gemm_tile<CIN>(xr  + (size_t)mr * CIN,        wroot + (size_t)(n0 + r) * CIN, half, c);

  // C/D layout: VGPR j -> row m0 + j + 8*half, col n0 + (lane & 15)
  const float bv = bias[n0 + r];
  #pragma unroll
  for (int j = 0; j < 8; ++j) {
    const int row = m0 + j + 8 * half;
    if (row < M) {
      float v = c[j] + bv;
      out[(size_t)row * Cout + (n0 + r)] = fmaxf(v, 0.0f);
    }
  }
}

// ---------------------------------------------------------------------------
// Edge scatter: agg[dst] += x[src], C channels, float4 per thread.
// For C=128 one wave covers exactly one edge (32 lanes x 4ch, coalesced 512B).
// ---------------------------------------------------------------------------
template<int C>
__global__ void edge_scatter(const float* __restrict__ x,
                             const int* __restrict__ src,
                             const int* __restrict__ dst,
                             float* __restrict__ agg,
                             int E)
{
  constexpr int CG = C / 4;
  const int t = blockIdx.x * blockDim.x + threadIdx.x;
  const int total = E * CG;
  if (t >= total) return;
  const int e  = t / CG;            // CG is a power of two -> shift
  const int cg = t - e * CG;
  const int s = src[e], d = dst[e];
  if (s < 0 || d < 0) return;       // invalid (pooled-away) edge
  const float4 v = *(const float4*)(x + (size_t)s * C + cg * 4);
  float* p = agg + (size_t)d * C + cg * 4;
  atomicAdd(p + 0, v.x);
  atomicAdd(p + 1, v.y);
  atomicAdd(p + 2, v.z);
  atomicAdd(p + 3, v.w);
}

// ---------------------------------------------------------------------------
// Utility fills
// ---------------------------------------------------------------------------
__global__ void fill_f32(float* p, int n, float v)
{ int i = blockIdx.x * blockDim.x + threadIdx.x; if (i < n) p[i] = v; }

__global__ void fill_i32(int* p, int n, int v)
{ int i = blockIdx.x * blockDim.x + threadIdx.x; if (i < n) p[i] = v; }

// ---------------------------------------------------------------------------
// TopK pooling: score = tanh((x.w)/||w||)
// ---------------------------------------------------------------------------
template<int C>
__global__ void score_kernel(const float* __restrict__ x,
                             const float* __restrict__ w,
                             float* __restrict__ score, int n)
{
  const int i = blockIdx.x * blockDim.x + threadIdx.x;
  if (i >= n) return;
  const float* row = x + (size_t)i * C;
  float nrm2 = 0.f, dot = 0.f;
  #pragma unroll 4
  for (int c = 0; c < C; ++c) { const float wc = w[c]; nrm2 += wc * wc; dot += row[c] * wc; }
  score[i] = tanhf(dot / sqrtf(nrm2));
}

__device__ inline unsigned f2ord(float f)
{
  unsigned u = __float_as_uint(f);
  return (u & 0x80000000u) ? ~u : (u | 0x80000000u);   // monotone: bigger float -> bigger uint
}

// 4-pass 8-bit radix-select of the k-th largest score. Single workgroup.
// sel[0] = threshold (orderable bits), sel[1] = #elements equal to threshold to take.
__global__ void select_kth_kernel(const float* __restrict__ score, int n, int k,
                                  int* __restrict__ sel)
{
  __shared__ unsigned hist[256];
  __shared__ unsigned s_prefix;
  __shared__ int      s_need;
  const int tid = threadIdx.x;
  if (tid == 0) { s_prefix = 0u; s_need = k; }
  __syncthreads();

  for (int pass = 0; pass < 4; ++pass) {
    const int shift = 24 - 8 * pass;
    for (int b = tid; b < 256; b += blockDim.x) hist[b] = 0u;
    __syncthreads();
    const unsigned prefix = s_prefix;
    const unsigned maskHi = (pass == 0) ? 0u : (0xFFFFFFFFu << (shift + 8));
    for (int i = tid; i < n; i += blockDim.x) {
      const unsigned u = f2ord(score[i]);
      if ((u & maskHi) == prefix)
        atomicAdd(&hist[(u >> shift) & 255u], 1u);
    }
    __syncthreads();
    if (tid == 0) {
      const int need = s_need;
      unsigned cum = 0;
      int b = 255;
      for (; b > 0; --b) {
        const unsigned c = hist[b];
        if (cum + c >= (unsigned)need) break;
        cum += c;
      }
      s_prefix = prefix | ((unsigned)b << shift);
      s_need   = need - (int)cum;
    }
    __syncthreads();
  }
  if (tid == 0) { sel[0] = (int)s_prefix; sel[1] = s_need; }
}

// Compact kept node indices: all score > T, plus (k - nG) of score == T.
__global__ void compact_kernel(const float* __restrict__ score, int n, int k,
                               const int* __restrict__ sel,
                               int* __restrict__ keep, int* __restrict__ counters)
{
  const unsigned T = (unsigned)sel[0];
  const int take_eq = sel[1];
  const int nG = k - take_eq;
  const int i = blockIdx.x * blockDim.x + threadIdx.x;
  if (i >= n) return;
  const unsigned u = f2ord(score[i]);
  if (u > T) {
    const int pos = atomicAdd(&counters[0], 1);
    keep[pos] = i;
  } else if (u == T) {
    const int pos = atomicAdd(&counters[1], 1);
    if (pos < take_eq) keep[nG + pos] = i;
  }
}

__global__ void map_set_kernel(int* __restrict__ map, const int* __restrict__ keep, int k)
{ int j = blockIdx.x * blockDim.x + threadIdx.x; if (j < k) map[keep[j]] = j; }

// xp[j] = x[keep[j]] * score[keep[j]]
template<int C>
__global__ void pool_gather(const float* __restrict__ x, const int* __restrict__ keep,
                            const float* __restrict__ score, float* __restrict__ xp, int k)
{
  const int t = blockIdx.x * blockDim.x + threadIdx.x;
  const int total = k * C;
  if (t >= total) return;
  const int j = t / C, c = t - j * C;
  const int s = keep[j];
  xp[t] = x[(size_t)s * C + c] * score[s];
}

// Remap edge endpoints through pooling map (invalid -> -1)
__global__ void edge_remap(const int* __restrict__ src_in, const int* __restrict__ dst_in,
                           const int* __restrict__ map,
                           int* __restrict__ src_out, int* __restrict__ dst_out, int E)
{
  const int e = blockIdx.x * blockDim.x + threadIdx.x;
  if (e >= E) return;
  const int s = src_in[e], d = dst_in[e];
  src_out[e] = (s >= 0) ? map[s] : -1;
  dst_out[e] = (d >= 0) ? map[d] : -1;
}

// u[i] = xbase[i] + (map[i] >= 0 ? xsmall[map[i]] : 0)     (unpool + skip add)
template<int C>
__global__ void unpool_add(const float* __restrict__ xbase, const float* __restrict__ xsmall,
                           const int* __restrict__ map, float* __restrict__ u, int n)
{
  const int t = blockIdx.x * blockDim.x + threadIdx.x;
  const int total = n * C;
  if (t >= total) return;
  const int i = t / C, c = t - i * C;
  float v = xbase[t];
  const int m = map[i];
  if (m >= 0) v += xsmall[(size_t)m * C + c];
  u[t] = v;
}

// ---------------------------------------------------------------------------
// Classifier: out = log_softmax(x @ lin_w^T + lin_b), NCLS = 10, FEAT = 128
// ---------------------------------------------------------------------------
__global__ void classifier_kernel(const float* __restrict__ x,
                                  const float* __restrict__ w,
                                  const float* __restrict__ b,
                                  float* __restrict__ out, int n)
{
  __shared__ float sw[10 * 128];
  __shared__ float sb[10];
  for (int t = threadIdx.x; t < 10 * 128; t += blockDim.x) sw[t] = w[t];
  if (threadIdx.x < 10) sb[threadIdx.x] = b[threadIdx.x];
  __syncthreads();

  const int i = blockIdx.x * blockDim.x + threadIdx.x;
  if (i >= n) return;
  const float* row = x + (size_t)i * 128;
  float acc[10];
  #pragma unroll
  for (int c = 0; c < 10; ++c) acc[c] = sb[c];
  for (int k = 0; k < 128; ++k) {
    const float xv = row[k];
    #pragma unroll
    for (int c = 0; c < 10; ++c) acc[c] += xv * sw[c * 128 + k];
  }
  float m = acc[0];
  #pragma unroll
  for (int c = 1; c < 10; ++c) m = fmaxf(m, acc[c]);
  float s = 0.f;
  #pragma unroll
  for (int c = 0; c < 10; ++c) s += expf(acc[c] - m);
  const float lse = logf(s);
  #pragma unroll
  for (int c = 0; c < 10; ++c) out[(size_t)i * 10 + c] = acc[c] - m - lse;
}

// ---------------------------------------------------------------------------
// Host orchestration
// ---------------------------------------------------------------------------
static inline int cdiv(int a, int b) { return (a + b - 1) / b; }

extern "C" void kernel_launch(void* const* d_in, const int* in_sizes, int n_in,
                              void* d_out, int out_size, void* d_ws, size_t ws_size,
                              hipStream_t stream)
{
  const float* x       = (const float*)d_in[0];
  const int*   ei      = (const int*)  d_in[1];
  const float* w1_rel  = (const float*)d_in[2];
  const float* w1_root = (const float*)d_in[3];
  const float* b1      = (const float*)d_in[4];
  const float* p1_w    = (const float*)d_in[5];
  const float* w2_rel  = (const float*)d_in[6];
  const float* w2_root = (const float*)d_in[7];
  const float* b2      = (const float*)d_in[8];
  const float* p2_w    = (const float*)d_in[9];
  const float* w3_rel  = (const float*)d_in[10];
  const float* w3_root = (const float*)d_in[11];
  const float* b3      = (const float*)d_in[12];
  const float* w4_rel  = (const float*)d_in[13];
  const float* w4_root = (const float*)d_in[14];
  const float* b4      = (const float*)d_in[15];
  const float* w5_rel  = (const float*)d_in[16];
  const float* w5_root = (const float*)d_in[17];
  const float* b5      = (const float*)d_in[18];
  const float* lin_w   = (const float*)d_in[19];
  const float* lin_b   = (const float*)d_in[20];

  const int N  = in_sizes[0] / 128;          // 50000
  const int E  = in_sizes[1] / 2;            // 1600000
  const int K1 = (4 * N + 4) / 5;            // ceil(0.8*N)  = 40000
  const int K2 = (4 * K1 + 4) / 5;           // ceil(0.8*K1) = 32000

  // --- workspace arena -----------------------------------------------------
  char* W = (char*)d_ws;
  size_t off = 0;
  auto alloc = [&](size_t bytes) -> void* {
    void* p = W + off;
    off += (bytes + 255) & ~(size_t)255;
    return p;
  };
  float* agg    = (float*)alloc((size_t)N  * 128 * 4);
  float* x1     = (float*)alloc((size_t)N  * 128 * 4);
  float* xp1    = (float*)alloc((size_t)K1 * 128 * 4);
  float* x2     = (float*)alloc((size_t)K1 * 64  * 4);
  float* xp2    = (float*)alloc((size_t)K2 * 64  * 4);
  float* x3     = (float*)alloc((size_t)K2 * 64  * 4);
  float* u1     = (float*)alloc((size_t)K1 * 64  * 4);
  float* x4     = (float*)alloc((size_t)K1 * 128 * 4);
  float* u2     = (float*)alloc((size_t)N  * 128 * 4);
  float* score1 = (float*)alloc((size_t)N  * 4);
  float* score2 = (float*)alloc((size_t)K1 * 4);
  int*   keep1  = (int*)  alloc((size_t)K1 * 4);
  int*   keep2  = (int*)  alloc((size_t)K2 * 4);
  int*   map1   = (int*)  alloc((size_t)N  * 4);
  int*   map2   = (int*)  alloc((size_t)K1 * 4);
  int*   src1   = (int*)  alloc((size_t)E * 4);
  int*   dst1   = (int*)  alloc((size_t)E * 4);
  int*   src2   = (int*)  alloc((size_t)E * 4);
  int*   dst2   = (int*)  alloc((size_t)E * 4);
  int*   sel1   = (int*)  alloc(64);
  int*   sel2   = (int*)  alloc(64);
  int*   cnt1   = (int*)  alloc(64);
  int*   cnt2   = (int*)  alloc(64);
  float* x5     = x1;     // x1 dead after unpool2 -> reuse for x5

  const int T = 256;
  const int* esrc = ei;
  const int* edst = ei + E;

  // ---- conv1: x1 = relu(gc(x; w1)) ---------------------------------------
  fill_f32<<<cdiv(N * 128, T), T, 0, stream>>>(agg, N * 128, 0.0f);
  edge_scatter<128><<<cdiv(E * 32, T), T, 0, stream>>>(x, esrc, edst, agg, E);
  graphconv_gemm<128><<<dim3(cdiv(N, 16), 8), 32, 0, stream>>>(
      agg, x, w1_rel, w1_root, b1, x1, N, 128);

  // ---- pool1 --------------------------------------------------------------
  score_kernel<128><<<cdiv(N, T), T, 0, stream>>>(x1, p1_w, score1, N);
  select_kth_kernel<<<1, 1024, 0, stream>>>(score1, N, K1, sel1);
  fill_i32<<<1, 64, 0, stream>>>(cnt1, 2, 0);
  compact_kernel<<<cdiv(N, T), T, 0, stream>>>(score1, N, K1, sel1, keep1, cnt1);
  fill_i32<<<cdiv(N, T), T, 0, stream>>>(map1, N, -1);
  map_set_kernel<<<cdiv(K1, T), T, 0, stream>>>(map1, keep1, K1);
  pool_gather<128><<<cdiv(K1 * 128, T), T, 0, stream>>>(x1, keep1, score1, xp1, K1);
  edge_remap<<<cdiv(E, T), T, 0, stream>>>(esrc, edst, map1, src1, dst1, E);

  // ---- conv2: x2 = relu(gc(xp1; w2)) -------------------------------------
  fill_f32<<<cdiv(K1 * 128, T), T, 0, stream>>>(agg, K1 * 128, 0.0f);
  edge_scatter<128><<<cdiv(E * 32, T), T, 0, stream>>>(xp1, src1, dst1, agg, E);
  graphconv_gemm<128><<<dim3(cdiv(K1, 16), 4), 32, 0, stream>>>(
      agg, xp1, w2_rel, w2_root, b2, x2, K1, 64);

  // ---- pool2 --------------------------------------------------------------
  score_kernel<64><<<cdiv(K1, T), T, 0, stream>>>(x2, p2_w, score2, K1);
  select_kth_kernel<<<1, 1024, 0, stream>>>(score2, K1, K2, sel2);
  fill_i32<<<1, 64, 0, stream>>>(cnt2, 2, 0);
  compact_kernel<<<cdiv(K1, T), T, 0, stream>>>(score2, K1, K2, sel2, keep2, cnt2);
  fill_i32<<<cdiv(K1, T), T, 0, stream>>>(map2, K1, -1);
  map_set_kernel<<<cdiv(K2, T), T, 0, stream>>>(map2, keep2, K2);
  pool_gather<64><<<cdiv(K2 * 64, T), T, 0, stream>>>(x2, keep2, score2, xp2, K2);
  edge_remap<<<cdiv(E, T), T, 0, stream>>>(src1, dst1, map2, src2, dst2, E);

  // ---- conv3: x3 = relu(gc(xp2; w3)) -------------------------------------
  fill_f32<<<cdiv(K2 * 64, T), T, 0, stream>>>(agg, K2 * 64, 0.0f);
  edge_scatter<64><<<cdiv(E * 16, T), T, 0, stream>>>(xp2, src2, dst2, agg, E);
  graphconv_gemm<64><<<dim3(cdiv(K2, 16), 4), 32, 0, stream>>>(
      agg, xp2, w3_rel, w3_root, b3, x3, K2, 64);

  // ---- unpool1 + skip: u1 = scatter(x3) + x2 ------------------------------
  unpool_add<64><<<cdiv(K1 * 64, T), T, 0, stream>>>(x2, x3, map2, u1, K1);

  // ---- conv4: x4 = relu(gc(u1; w4)) --------------------------------------
  fill_f32<<<cdiv(K1 * 64, T), T, 0, stream>>>(agg, K1 * 64, 0.0f);
  edge_scatter<64><<<cdiv(E * 16, T), T, 0, stream>>>(u1, src1, dst1, agg, E);
  graphconv_gemm<64><<<dim3(cdiv(K1, 16), 8), 32, 0, stream>>>(
      agg, u1, w4_rel, w4_root, b4, x4, K1, 128);

  // ---- unpool2 + skip: u2 = scatter(x4) + x1 ------------------------------
  unpool_add<128><<<cdiv(N * 128, T), T, 0, stream>>>(x1, x4, map1, u2, N);

  // ---- conv5: x5 = relu(gc(u2; w5)) --------------------------------------
  fill_f32<<<cdiv(N * 128, T), T, 0, stream>>>(agg, N * 128, 0.0f);
  edge_scatter<128><<<cdiv(E * 32, T), T, 0, stream>>>(u2, esrc, edst, agg, E);
  graphconv_gemm<128><<<dim3(cdiv(N, 16), 8), 32, 0, stream>>>(
      agg, u2, w5_rel, w5_root, b5, x5, N, 128);

  // ---- classifier + log_softmax ------------------------------------------
  classifier_kernel<<<cdiv(N, T), T, 0, stream>>>(x5, lin_w, lin_b, (float*)d_out, N);
}
